// SparseAutoencoder_7267084665348
// MI455X (gfx1250) — matrix operand order repaired and checked by reading, not compiled, not run on hardware
//
#include <hip/hip_runtime.h>
#include <hip/hip_bf16.h>

typedef __attribute__((ext_vector_type(16))) __bf16 v16bf;
typedef __attribute__((ext_vector_type(8)))  __bf16 v8bf;
typedef __attribute__((ext_vector_type(4)))  __bf16 v4bf;
typedef __attribute__((ext_vector_type(8)))  float  v8f;

#define D_IN    768
#define NCHUNK  (D_IN / 32)  // 24 K-chunks of 32
#define TOPK    32
#define HSPLIT  4
#define ROWS    32           // rows per wave: 2 WMMA M-tiles share every B fragment
#define LDSW    (D_IN + 8)   // +8 halves padding -> row stride 1552B, spreads LDS banks

// ---------------- Pass 0: W_enc f32 -> bf16 ----------------
__global__ __launch_bounds__(256) void convert_w_kernel(const float* __restrict__ W,
                                                        __bf16* __restrict__ Wb, int n) {
    int i = (blockIdx.x * 256 + threadIdx.x) * 4;
    if (i + 3 < n) {
        float4 f = *(const float4*)(W + i);
        v4bf o = { (__bf16)f.x, (__bf16)f.y, (__bf16)f.z, (__bf16)f.w };
        *(v4bf*)(Wb + i) = o;
    }
}

// Issue async copies for one 16-row B tile (16 x 768 bf16 = 48 x 512B) into LDS.
// Tracked on ASYNCcnt; no data VGPRs consumed.
__device__ __forceinline__ void async_b_tile(const __bf16* __restrict__ Wb, int hrow0,
                                             unsigned lds_byte_base, int lane) {
    #pragma unroll
    for (int r = 0; r < 16; ++r) {
        const char* g = (const char*)(Wb + (size_t)(hrow0 + r) * D_IN) + lane * 16;
        unsigned    l = lds_byte_base + (unsigned)(r * (LDSW * 2)) + (unsigned)(lane * 16);
        #pragma unroll
        for (int j = 0; j < 3; ++j) {   // 3 * 32 lanes * 16B = 1536B = one W row
            asm volatile("global_load_async_to_lds_b128 %0, %1, off"
                         :: "v"(l + j * 512), "v"(g + j * 512) : "memory");
        }
    }
}

// ---------------- Pass 1: fused encode (WMMA bf16, async-LDS B pipeline) + top-k ------
// 1 wave per block; block (bx,by): rows [bx*32, bx*32+32), features [by*PH, (by+1)*PH)
__global__ __launch_bounds__(32) void encode_topk_kernel(
    const float* __restrict__ x, const __bf16* __restrict__ Wb,
    const float* __restrict__ b_enc,
    float* __restrict__ cv, int* __restrict__ ci, int d_hidden)
{
    __shared__ __attribute__((aligned(16))) __bf16 xlds[ROWS * LDSW];
    __shared__ __attribute__((aligned(16))) __bf16 blds[2 * 16 * LDSW];  // double-buffered B tile
    __shared__ float  sval[ROWS][TOPK];
    __shared__ int    sidx[ROWS][TOPK];
    __shared__ float  sthr[ROWS];
    __shared__ int    smin[ROWS];

    const int lane = threadIdx.x;
    const int r0   = blockIdx.x * ROWS;
    const int part = blockIdx.y;
    const int PH   = d_hidden / HSPLIT;
    const int hlo  = part * PH;

    // init running top-k state (val=0 matches ReLU zero-floor semantics)
    #pragma unroll
    for (int row = 0; row < ROWS; ++row) { sval[row][lane] = 0.0f; sidx[row][lane] = 0; }
    sthr[lane] = 0.0f;            // lane 0..31 == ROWS
    smin[lane] = 0;

    // load & convert this wave's 32x768 x-tile into LDS (bf16)
    for (int i = lane; i < ROWS * D_IN; i += 32) {
        int m = i / D_IN, d = i - m * D_IN;
        xlds[m * LDSW + d] = (__bf16)x[(size_t)(r0 + m) * D_IN + d];
    }

    const int n = lane & 15;     // WMMA N (and A's M within a tile) for this lane
    const int g = lane >> 4;     // lane half-group

    // Preload ALL A fragments into registers (h-invariant; 2 tiles x 24 chunks x 8 VGPR).
    // Held as SSA values so the asm memory clobbers below cannot force reloads.
    v16bf afrag0[NCHUNK], afrag1[NCHUNK];
    #pragma unroll
    for (int c = 0; c < NCHUNK; ++c) {
        v8bf l0 = *(const v8bf*)&xlds[n * LDSW + c * 32 + g * 8];
        v8bf h0 = *(const v8bf*)&xlds[n * LDSW + c * 32 + 16 + g * 8];
        afrag0[c] = __builtin_shufflevector(l0, h0, 0,1,2,3,4,5,6,7,8,9,10,11,12,13,14,15);
        v8bf l1 = *(const v8bf*)&xlds[(16 + n) * LDSW + c * 32 + g * 8];
        v8bf h1 = *(const v8bf*)&xlds[(16 + n) * LDSW + c * 32 + 16 + g * 8];
        afrag1[c] = __builtin_shufflevector(l1, h1, 0,1,2,3,4,5,6,7,8,9,10,11,12,13,14,15);
    }

    const unsigned blds_base = (unsigned)(uintptr_t)&blds[0];
    const unsigned bufstride = (unsigned)(16 * LDSW * 2);   // bytes per B buffer

    // prologue: start tile 0 into buffer 0
    async_b_tile(Wb, hlo, blds_base, lane);

    for (int ht = 0; ht < PH; ht += 16) {
        const int h0  = hlo + ht;
        const int cur = (ht >> 4) & 1;

        // wait for current tile's async copies, then start the next tile
        asm volatile("s_wait_asynccnt 0" ::: "memory");
        if (ht + 16 < PH)
            async_b_tile(Wb, h0 + 16, blds_base + (cur ^ 1) * bufstride, lane);
        __builtin_prefetch((const void*)(Wb + (size_t)(h0 + 32) * D_IN + lane * 8), 0, 3);

        const int bb = cur * 16 * LDSW + n * LDSW + g * 8;  // half-index of this lane's B row slice
        v8f acc0 = {0.f,0.f,0.f,0.f,0.f,0.f,0.f,0.f};
        v8f acc1 = {0.f,0.f,0.f,0.f,0.f,0.f,0.f,0.f};
        #pragma unroll
        for (int c = 0; c < NCHUNK; ++c) {
            // B fragment: W_enc[h0+n, K=c*32 + g*8 + j] (j<8), K=c*32 + 16 + g*8 + (j-8) (j>=8) from LDS
            v8bf b0 = *(const v8bf*)&blds[bb + c * 32];
            v8bf b1 = *(const v8bf*)&blds[bb + c * 32 + 16];
            v16bf b = __builtin_shufflevector(b0, b1, 0,1,2,3,4,5,6,7,8,9,10,11,12,13,14,15);
            acc0 = __builtin_amdgcn_wmma_f32_16x16x32_bf16(false, afrag0[c], false, b,
                                                           (short)0, acc0, false, false);
            acc1 = __builtin_amdgcn_wmma_f32_16x16x32_bf16(false, afrag1[c], false, b,
                                                           (short)0, acc1, false, false);
        }

        const float bias = b_enc[h0 + n];
        #pragma unroll
        for (int t = 0; t < 2; ++t) {
            #pragma unroll
            for (int r = 0; r < 8; ++r) {
                float v  = (t == 0 ? acc0[r] : acc1[r]) + bias; // C: M = r + 8*g, N = n
                int  row = t * 16 + r + 8 * g;
                float thr = sthr[row];          // thr >= 0 always => v>thr implies ReLU-pass
                unsigned mask = (unsigned)__ballot(v > thr);
                while (mask) {                  // rare path: candidate beats current 32nd
                    int src = __builtin_ctz(mask);
                    mask &= mask - 1;
                    float bv  = __shfl(v,   src, 32);
                    int  brow = __shfl(row, src, 32);
                    int  bh   = h0 + __shfl(n, src, 32);
                    if (bv > sthr[brow]) {
                        if (lane == 0) {
                            int p = smin[brow];
                            sval[brow][p] = bv;
                            sidx[brow][p] = bh;
                        }
                        float e  = sval[brow][lane];    // one LDS read per lane
                        float mv = e;
                        #pragma unroll
                        for (int off = 16; off > 0; off >>= 1)
                            mv = fminf(mv, __shfl_xor(mv, off, 32));
                        unsigned pm = (unsigned)__ballot(e == mv);
                        if (lane == 0) { sthr[brow] = mv; smin[brow] = __builtin_ctz(pm); }
                    }
                }
            }
        }
    }

    // emit partial top-32 candidates for this feature partition
    #pragma unroll
    for (int row = 0; row < ROWS; ++row) {
        size_t o = ((size_t)(r0 + row) * HSPLIT + part) * TOPK + lane;
        cv[o] = sval[row][lane];
        ci[o] = sidx[row][lane];
    }
}

// ---------------- Pass 2: merge HSPLIT partial lists -> exact top-32 per row ----------
__global__ __launch_bounds__(32) void merge_topk_kernel(
    const float* __restrict__ cv, const int* __restrict__ ci,
    float* __restrict__ tv, int* __restrict__ ti)
{
    const int row = blockIdx.x, lane = threadIdx.x;
    float v[HSPLIT]; int id[HSPLIT];
    #pragma unroll
    for (int j = 0; j < HSPLIT; ++j) {
        size_t o = (size_t)row * (HSPLIT * TOPK) + j * 32 + lane;
        v[j] = cv[o]; id[j] = ci[o];
    }
    for (int s = 0; s < TOPK; ++s) {
        float bm = v[0]; int bj = 0;
        #pragma unroll
        for (int j = 1; j < HSPLIT; ++j) if (v[j] > bm) { bm = v[j]; bj = j; }
        float rv = bm; int rl = lane;
        #pragma unroll
        for (int off = 16; off > 0; off >>= 1) {
            float ov = __shfl_xor(rv, off, 32);
            int   ol = __shfl_xor(rl, off, 32);
            if (ov > rv || (ov == rv && ol < rl)) { rv = ov; rl = ol; }
        }
        int myidx = id[bj];
        int ridx  = __shfl(myidx, rl, 32);
        if (lane == 0) { tv[(size_t)row * TOPK + s] = rv; ti[(size_t)row * TOPK + s] = ridx; }
        if (lane == rl) {
            #pragma unroll
            for (int j = 0; j < HSPLIT; ++j) if (j == bj) v[j] = -1.0f;
        }
    }
}

// ---------------- Pass 3: sparse decode (fp32, W_enc rows from L2) ----------------
__global__ __launch_bounds__(256) void decode_kernel(
    const float* __restrict__ W, const float* __restrict__ b_dec,
    const float* __restrict__ tv, const int* __restrict__ ti,
    float* __restrict__ out)
{
    const int row = blockIdx.x;
    __shared__ float vv[TOPK];
    __shared__ int   hh[TOPK];
    if (threadIdx.x < TOPK) {
        vv[threadIdx.x] = tv[(size_t)row * TOPK + threadIdx.x];
        hh[threadIdx.x] = ti[(size_t)row * TOPK + threadIdx.x];
    }
    __syncthreads();
    const int c0 = threadIdx.x, c1 = threadIdx.x + 256, c2 = threadIdx.x + 512;
    float a0 = 0.f, a1 = 0.f, a2 = 0.f;
    #pragma unroll 4
    for (int j = 0; j < TOPK; ++j) {
        float vj = vv[j];
        const float* wrow = W + (size_t)hh[j] * D_IN;
        a0 += vj * wrow[c0]; a1 += vj * wrow[c1]; a2 += vj * wrow[c2];
    }
    out[(size_t)row * D_IN + c0] = a0 + b_dec[c0];
    out[(size_t)row * D_IN + c1] = a1 + b_dec[c1];
    out[(size_t)row * D_IN + c2] = a2 + b_dec[c2];
}

// ---------------- Launch ----------------
extern "C" void kernel_launch(void* const* d_in, const int* in_sizes, int n_in,
                              void* d_out, int out_size, void* d_ws, size_t ws_size,
                              hipStream_t stream) {
    const float* x     = (const float*)d_in[0];
    const float* W     = (const float*)d_in[1];
    const float* b_enc = (const float*)d_in[2];
    const float* b_dec = (const float*)d_in[3];
    const int d_hidden = in_sizes[2];                // 32768
    const int Nrows    = in_sizes[0] / D_IN;         // 8192
    const size_t wcnt  = (size_t)d_hidden * D_IN;

    // workspace layout
    char* p = (char*)d_ws;
    __bf16* Wb = (__bf16*)p;            p += wcnt * sizeof(__bf16);                 // 48 MB
    float*  cv = (float*)p;             p += (size_t)Nrows * HSPLIT * TOPK * 4;     // 4 MB
    int*    ci = (int*)p;               p += (size_t)Nrows * HSPLIT * TOPK * 4;     // 4 MB
    float*  tv = (float*)p;             p += (size_t)Nrows * TOPK * 4;              // 1 MB
    int*    ti = (int*)p;

    int cblocks = (int)((wcnt / 4 + 255) / 256);
    convert_w_kernel<<<cblocks, 256, 0, stream>>>(W, Wb, (int)wcnt);

    dim3 eg(Nrows / ROWS, HSPLIT);
    encode_topk_kernel<<<eg, 32, 0, stream>>>(x, Wb, b_enc, cv, ci, d_hidden);

    merge_topk_kernel<<<Nrows, 32, 0, stream>>>(cv, ci, tv, ti);

    decode_kernel<<<Nrows, 256, 0, stream>>>(W, b_dec, tv, ti, (float*)d_out);
}
